// TransformerEncoder_81363860455721
// MI455X (gfx1250) — matrix operand order repaired and checked by reading, not compiled, not run on hardware
//
#include <hip/hip_runtime.h>
#include <cstdint>

#define DEV __device__ __forceinline__

typedef __attribute__((ext_vector_type(16))) __bf16 bf16x16;
typedef __attribute__((ext_vector_type(8)))  float  f32x8;
typedef unsigned int u32x4 __attribute__((ext_vector_type(4)));
typedef unsigned int u32x8 __attribute__((ext_vector_type(8)));

union FragAB {
  bf16x16        v;
  unsigned short h[16];
  unsigned int   w[8];
};

DEV unsigned short f2bf(float f) {
  unsigned u = __float_as_uint(f);
  unsigned r = u + 0x7FFFu + ((u >> 16) & 1u);   // round-to-nearest-even
  return (unsigned short)(r >> 16);
}
DEV float bf2f(unsigned short h) { return __uint_as_float(((unsigned)h) << 16); }

// ---------------------------------------------------------------------------
// Tensor Data Mover: async 2D bf16 tile load global->LDS with per-row pad.
// Builds D# group0 (4 SGPRs) + group1 (8 SGPRs); VADDR2/3 = NULL (2D form).
// pad_int_code: 0=2,1=4,2=8,3=16,4=32... dwords between pads; pad = 1 dword.
// ---------------------------------------------------------------------------
DEV void tdm_load_2d_bf16(unsigned lds_off, const unsigned short* g,
                          unsigned tensor_w, unsigned tensor_h,
                          unsigned stride_elems, unsigned tile_w,
                          unsigned tile_h, unsigned pad_int_code) {
  unsigned long long ga = (unsigned long long)(uintptr_t)g;
  u32x4 g0;
  g0[0] = 1u;                                   // count=1, user mode
  g0[1] = lds_off;                              // lds_addr (bytes)
  g0[2] = (unsigned)ga;                         // global_addr[31:0]
  g0[3] = (unsigned)((ga >> 32) & 0x01FFFFFFu)  // global_addr[56:32]
          | (2u << 30);                         // type = 2 ("image")
  u32x8 g1;
  g1[0] = (1u << 16)                            // data_size = 2 bytes
          | (1u << 20)                          // pad_enable
          | (pad_int_code << 22)                // pad_interval
          | (0u << 25);                         // pad_amount = 1 dword
  g1[1] = (tensor_w & 0xFFFFu) << 16;           // tensor_dim0[15:0]
  g1[2] = (tensor_w >> 16) | ((tensor_h & 0xFFFFu) << 16);
  g1[3] = (tensor_h >> 16) | (tile_w << 16);    // tile_dim0
  g1[4] = tile_h;                               // tile_dim1 (tile_dim2 = 0)
  g1[5] = stride_elems;                         // tensor_dim0_stride[31:0]
  g1[6] = 0;
  g1[7] = 0;
  asm volatile("tensor_load_to_lds %0, %1" :: "s"(g0), "s"(g1) : "memory");
}

// ---------------------------------------------------------------------------
__global__ void cvt_f32_bf16(const float* __restrict__ in,
                             unsigned short* __restrict__ out, int n) {
  int i = blockIdx.x * blockDim.x + threadIdx.x;
  int stride = gridDim.x * blockDim.x;
  for (; i < n; i += stride) out[i] = f2bf(in[i]);
}

__global__ void init_h(const float* __restrict__ x, float* __restrict__ h,
                       unsigned short* __restrict__ hb, int n) {
  int i = blockIdx.x * blockDim.x + threadIdx.x;
  int stride = gridDim.x * blockDim.x;
  for (; i < n; i += stride) {
    float v = x[i];
    h[i] = v;
    hb[i] = f2bf(v);
  }
}

// ---------------------------------------------------------------------------
// C = act(A[MxK] @ W[KxN] + bias). bf16 in, f32 WMMA accumulate.
// Block tile 128x64, 8 waves, each wave a 32x32 patch (2x2 frags).
// A tile loaded by TDM (row-major + pad); B tile staged transposed by VALU.
// ---------------------------------------------------------------------------
#define BM 128
#define BN 64
#define BK 32
#define AS_LD 34   // 32 + 2 pad (halfwords)
#define KS_LD 66   // 64 + 2 pad (halfwords)

__global__ __launch_bounds__(256) void gemm_bf16_wmma(
    const unsigned short* __restrict__ A, const unsigned short* __restrict__ W,
    const float* __restrict__ bias, float* __restrict__ Cf,
    unsigned short* __restrict__ Cb, int M, int N, int K, int relu) {
  __shared__ unsigned short As[BM * AS_LD];   // [row][k], TDM-filled
  __shared__ unsigned short Bs[BN * AS_LD];   // [col][k], transposed

  const int tid  = threadIdx.x;
  const int lane = tid & 31;
  const int wave = tid >> 5;
  const int ln   = lane & 15;
  const int half = lane >> 4;
  const int wr   = wave >> 1;
  const int wc   = wave & 1;
  const int rowBase = blockIdx.y * BM;
  const int colBase = blockIdx.x * BN;

  const unsigned asOff = (unsigned)(uintptr_t)(&As[0]);

  f32x8 c[2][2];
#pragma unroll
  for (int mi = 0; mi < 2; ++mi)
#pragma unroll
    for (int ni = 0; ni < 2; ++ni)
#pragma unroll
      for (int e = 0; e < 8; ++e) c[mi][ni][e] = 0.0f;

  const unsigned int* W32 = (const unsigned int*)W;
  unsigned int* Bs32 = (unsigned int*)Bs;

  for (int kt = 0; kt < K; kt += BK) {
    // async DMA of the 128x32 A tile into padded LDS (one issue per block)
    if (wave == 0)
      tdm_load_2d_bf16(asOff, A + (size_t)rowBase * K + kt,
                       (unsigned)K, (unsigned)M, (unsigned)K, BK, BM,
                       /*pad every 16 dwords*/ 3);

    // stage B tile transposed: Bs[col][k]; 512 2x2 units, 2 per thread
#pragma unroll
    for (int u = 0; u < 2; ++u) {
      int uu = tid + 256 * u;            // 0..511
      int rp = uu >> 5, cp = uu & 31;    // row pair, col pair
      size_t base = ((size_t)(kt + 2 * rp) * N + colBase) / 2 + cp;
      unsigned w0 = W32[base];
      unsigned w1 = W32[base + (N >> 1)];
      Bs32[cp * 34 + rp]      = (w0 & 0xFFFFu) | (w1 << 16);
      Bs32[cp * 34 + 17 + rp] = (w0 >> 16) | (w1 & 0xFFFF0000u);
    }
    if (kt + BK < K)   // hint next W tile (global_prefetch_b8)
      __builtin_prefetch(&W[(size_t)(kt + BK + (tid >> 3)) * N + colBase], 0, 3);
    if (wave == 0) __builtin_amdgcn_s_wait_tensorcnt(0);
    __syncthreads();

    FragAB a[2], b[2];
#pragma unroll
    for (int mi = 0; mi < 2; ++mi) {
      int row = wr * 32 + mi * 16 + ln;
#pragma unroll
      for (int j = 0; j < 8; ++j) {
        int kk = (j >> 2) * 16 + half * 8 + 2 * (j & 3);
        a[mi].w[j] = *(const unsigned int*)&As[row * AS_LD + kk];
      }
    }
#pragma unroll
    for (int ni = 0; ni < 2; ++ni) {
      int col = wc * 32 + ni * 16 + ln;
#pragma unroll
      for (int j = 0; j < 8; ++j)
        b[ni].w[j] = *(const unsigned int*)&Bs[col * AS_LD + (j >> 2) * 16 + half * 8 + 2 * (j & 3)];
    }
#pragma unroll
    for (int mi = 0; mi < 2; ++mi)
#pragma unroll
      for (int ni = 0; ni < 2; ++ni)
        c[mi][ni] = __builtin_amdgcn_wmma_f32_16x16x32_bf16(
            false, a[mi].v, false, b[ni].v, (short)0, c[mi][ni], false, false);
    __syncthreads();
  }

#pragma unroll
  for (int mi = 0; mi < 2; ++mi) {
#pragma unroll
    for (int ni = 0; ni < 2; ++ni) {
      int col = colBase + wc * 32 + ni * 16 + ln;
      float bv = bias[col];
#pragma unroll
      for (int r = 0; r < 8; ++r) {
        int row = rowBase + wr * 32 + mi * 16 + r + 8 * half;
        float v = c[mi][ni][r] + bv;
        if (relu) v = fmaxf(v, 0.0f);
        size_t idx = (size_t)row * N + col;
        if (Cf) Cf[idx] = v;
        if (Cb) Cb[idx] = f2bf(v);
      }
    }
  }
}

// ---------------------------------------------------------------------------
// Flash attention. Q,K,V bf16 [B,S,D]; head h = cols h*64..h*64+63.
// 4 waves/block; each wave owns 16 query rows; K tile via TDM, V transposed.
// ---------------------------------------------------------------------------
__global__ __launch_bounds__(128) void flash_attn(
    const unsigned short* __restrict__ Q, const unsigned short* __restrict__ Km,
    const unsigned short* __restrict__ V, const float* __restrict__ mask,
    unsigned short* __restrict__ O, int B, int S, int Dm, int H) {
  const int tid  = threadIdx.x;
  const int lane = tid & 31;
  const int wave = tid >> 5;
  const int ln   = lane & 15;
  const int half = lane >> 4;

  const int bh   = blockIdx.y;
  const int b    = bh / H;
  const int hh   = bh % H;
  const int col0 = hh * 64;
  const int qRow0 = blockIdx.x * 64 + wave * 16;

  __shared__ unsigned short Ks[32 * KS_LD];   // [key][dh], TDM-filled
  __shared__ unsigned short Vs[64 * AS_LD];   // [dh][key], transposed
  __shared__ unsigned short Ps[4][16 * AS_LD];

  const unsigned ksOff = (unsigned)(uintptr_t)(&Ks[0]);
  const float invScale = 1.0f / 32.0f;        // 1/sqrt(d_model)

  FragAB qf[2];
#pragma unroll
  for (int ch = 0; ch < 2; ++ch) {
    const size_t rbase = ((size_t)(b * S + qRow0 + ln)) * Dm + col0;
#pragma unroll
    for (int j = 0; j < 8; ++j) {
      int dk = ch * 32 + (j >> 2) * 16 + half * 8 + 2 * (j & 3);
      unsigned int wv = *(const unsigned int*)&Q[rbase + dk];
      qf[ch].h[2 * j]     = f2bf(bf2f((unsigned short)(wv & 0xFFFFu)) * invScale);
      qf[ch].h[2 * j + 1] = f2bf(bf2f((unsigned short)(wv >> 16)) * invScale);
    }
  }

  f32x8 o[4];
#pragma unroll
  for (int c2 = 0; c2 < 4; ++c2)
#pragma unroll
    for (int e = 0; e < 8; ++e) o[c2][e] = 0.0f;
  float mrow[8], lrow[8];
#pragma unroll
  for (int r = 0; r < 8; ++r) { mrow[r] = -1e30f; lrow[r] = 0.0f; }

  const unsigned int* V32 = (const unsigned int*)V;
  unsigned int* Vs32 = (unsigned int*)Vs;

  for (int kb0 = 0; kb0 < S; kb0 += 32) {
    // async DMA of K tile (32 keys x 64 dh) into padded LDS
    if (wave == 0)
      tdm_load_2d_bf16(ksOff, Km + ((size_t)(b * S + kb0)) * Dm + col0,
                       (unsigned)Dm, (unsigned)(B * S), (unsigned)Dm, 64, 32,
                       /*pad every 32 dwords*/ 4);

    // stage V transposed: Vs[dh][key]; 512 2x2 units, 4 per thread
#pragma unroll
    for (int u = 0; u < 4; ++u) {
      int uu = tid + 128 * u;            // 0..511
      int kp = uu >> 5, dp = uu & 31;    // key pair, dh pair
      size_t base = (((size_t)(b * S + kb0 + 2 * kp)) * Dm + col0) / 2 + dp;
      unsigned w0 = V32[base];
      unsigned w1 = V32[base + (Dm >> 1)];
      Vs32[dp * 34 + kp]      = (w0 & 0xFFFFu) | (w1 << 16);
      Vs32[dp * 34 + 17 + kp] = (w0 >> 16) | (w1 & 0xFFFF0000u);
    }
    if (wave == 0) __builtin_amdgcn_s_wait_tensorcnt(0);
    __syncthreads();

    // S = Q * K^T
    f32x8 s[2];
#pragma unroll
    for (int n = 0; n < 2; ++n) {
#pragma unroll
      for (int e = 0; e < 8; ++e) s[n][e] = 0.0f;
#pragma unroll
      for (int ch = 0; ch < 2; ++ch) {
        FragAB kt;
#pragma unroll
        for (int j = 0; j < 8; ++j)
          kt.w[j] = *(const unsigned int*)
              &Ks[(n * 16 + ln) * KS_LD + ch * 32 + (j >> 2) * 16 + half * 8 + 2 * (j & 3)];
        s[n] = __builtin_amdgcn_wmma_f32_16x16x32_bf16(
            false, qf[ch].v, false, kt.v, (short)0, s[n], false, false);
      }
      float mv = mask[b * S + kb0 + n * 16 + ln] * (-1e9f);
#pragma unroll
      for (int e = 0; e < 8; ++e) s[n][e] += mv;
    }

    // online softmax (row reductions inside 16-lane halves)
    float rmax[8], rsum[8], fsc[8], mnew[8];
#pragma unroll
    for (int r = 0; r < 8; ++r) rmax[r] = fmaxf(s[0][r], s[1][r]);
#pragma unroll
    for (int r = 0; r < 8; ++r) {
#pragma unroll
      for (int off = 1; off < 16; off <<= 1)
        rmax[r] = fmaxf(rmax[r], __shfl_xor(rmax[r], off, 16));
      mnew[r] = fmaxf(mrow[r], rmax[r]);
      fsc[r]  = __expf(mrow[r] - mnew[r]);
      mrow[r] = mnew[r];
    }
#pragma unroll
    for (int r = 0; r < 8; ++r) rsum[r] = 0.0f;
#pragma unroll
    for (int n = 0; n < 2; ++n) {
#pragma unroll
      for (int r = 0; r < 8; ++r) {
        float p = __expf(s[n][r] - mnew[r]);
        rsum[r] += p;
        Ps[wave][(r + 8 * half) * AS_LD + n * 16 + ln] = f2bf(p);
      }
    }
#pragma unroll
    for (int r = 0; r < 8; ++r) {
#pragma unroll
      for (int off = 1; off < 16; off <<= 1)
        rsum[r] += __shfl_xor(rsum[r], off, 16);
      lrow[r] = lrow[r] * fsc[r] + rsum[r];
    }
#pragma unroll
    for (int c2 = 0; c2 < 4; ++c2)
#pragma unroll
      for (int r = 0; r < 8; ++r) o[c2][r] *= fsc[r];

    // repack P -> A-fragment layout via per-wave LDS
    FragAB pf;
#pragma unroll
    for (int j = 0; j < 8; ++j) {
      int kk = (j >> 2) * 16 + half * 8 + 2 * (j & 3);
      pf.w[j] = *(const unsigned int*)&Ps[wave][ln * AS_LD + kk];
    }
    // O += P * V  (Vs transposed -> contiguous fragment dwords)
#pragma unroll
    for (int c2 = 0; c2 < 4; ++c2) {
      FragAB vf;
#pragma unroll
      for (int j = 0; j < 8; ++j)
        vf.w[j] = *(const unsigned int*)
            &Vs[(c2 * 16 + ln) * AS_LD + (j >> 2) * 16 + half * 8 + 2 * (j & 3)];
      o[c2] = __builtin_amdgcn_wmma_f32_16x16x32_bf16(
          false, pf.v, false, vf.v, (short)0, o[c2], false, false);
    }
    __syncthreads();
  }

#pragma unroll
  for (int r = 0; r < 8; ++r) lrow[r] = 1.0f / lrow[r];
#pragma unroll
  for (int c2 = 0; c2 < 4; ++c2) {
#pragma unroll
    for (int r = 0; r < 8; ++r) {
      int row = qRow0 + r + 8 * half;
      O[((size_t)(b * S + row)) * Dm + col0 + c2 * 16 + ln] =
          f2bf(o[c2][r] * lrow[r]);
    }
  }
}

// ---------------------------------------------------------------------------
// h = LayerNorm(h + y) * g + b ; writes f32 (and bf16 copy). One block/row.
// ---------------------------------------------------------------------------
__global__ __launch_bounds__(256) void resid_ln(
    const float* __restrict__ h, const float* __restrict__ y,
    const float* __restrict__ g, const float* __restrict__ bta,
    float* __restrict__ hout, unsigned short* __restrict__ hb, int D) {
  __shared__ float red[256];
  int row = blockIdx.x;
  const float* hr = h + (size_t)row * D;
  const float* yr = y + (size_t)row * D;
  float vals[4];
  float sum = 0.0f;
#pragma unroll
  for (int i = 0; i < 4; ++i) {
    int col = threadIdx.x + i * 256;
    vals[i] = hr[col] + yr[col];
    sum += vals[i];
  }
  red[threadIdx.x] = sum;
  __syncthreads();
  for (int s2 = 128; s2 > 0; s2 >>= 1) {
    if (threadIdx.x < s2) red[threadIdx.x] += red[threadIdx.x + s2];
    __syncthreads();
  }
  float mean = red[0] / D;
  __syncthreads();
  float vsum = 0.0f;
#pragma unroll
  for (int i = 0; i < 4; ++i) { float d = vals[i] - mean; vsum += d * d; }
  red[threadIdx.x] = vsum;
  __syncthreads();
  for (int s2 = 128; s2 > 0; s2 >>= 1) {
    if (threadIdx.x < s2) red[threadIdx.x] += red[threadIdx.x + s2];
    __syncthreads();
  }
  float rstd = rsqrtf(red[0] / D + 1e-5f);
#pragma unroll
  for (int i = 0; i < 4; ++i) {
    int col = threadIdx.x + i * 256;
    float v = (vals[i] - mean) * rstd * g[col] + bta[col];
    hout[(size_t)row * D + col] = v;
    if (hb) hb[(size_t)row * D + col] = f2bf(v);
  }
}

// ---------------------------------------------------------------------------
extern "C" void kernel_launch(void* const* d_in, const int* in_sizes, int n_in,
                              void* d_out, int out_size, void* d_ws,
                              size_t ws_size, hipStream_t stream) {
  (void)in_sizes; (void)n_in; (void)out_size; (void)ws_size;
  const int L = 6, Dm = 1024, H = 16, FF = 4096, B = 2, S = 2048;
  const int M = B * S;

  const float* x    = (const float*)d_in[0];
  const float* mask = (const float*)d_in[1];
  const float* Wq = (const float*)d_in[2];
  const float* bq = (const float*)d_in[3];
  const float* Wk = (const float*)d_in[4];
  const float* bk = (const float*)d_in[5];
  const float* Wv = (const float*)d_in[6];
  const float* bv = (const float*)d_in[7];
  const float* Wo = (const float*)d_in[8];
  const float* bo = (const float*)d_in[9];
  const float* W1 = (const float*)d_in[10];
  const float* b1 = (const float*)d_in[11];
  const float* W2 = (const float*)d_in[12];
  const float* b2 = (const float*)d_in[13];
  const float* g1 = (const float*)d_in[14];
  const float* be1 = (const float*)d_in[15];
  const float* g2 = (const float*)d_in[16];
  const float* be2 = (const float*)d_in[17];

  float* h = (float*)d_out;   // hidden state lives in d_out across layers

  char* ws = (char*)d_ws;
  size_t off = 0;
  auto alloc = [&](size_t bytes) {
    char* p = ws + off;
    off += (bytes + 255) & ~(size_t)255;
    return p;
  };
  unsigned short* hb  = (unsigned short*)alloc((size_t)M * Dm * 2);
  unsigned short* qb  = (unsigned short*)alloc((size_t)M * Dm * 2);
  unsigned short* kb  = (unsigned short*)alloc((size_t)M * Dm * 2);
  unsigned short* vb  = (unsigned short*)alloc((size_t)M * Dm * 2);
  unsigned short* aob = (unsigned short*)alloc((size_t)M * Dm * 2);
  float*          tf  = (float*)alloc((size_t)M * Dm * 4);
  unsigned short* f1b = (unsigned short*)alloc((size_t)M * FF * 2);
  unsigned short* wb  = (unsigned short*)alloc((size_t)Dm * FF * 2);

  init_h<<<2048, 256, 0, stream>>>(x, h, hb, M * Dm);

  const dim3 gProj(Dm / BN, M / BM);
  const dim3 gFF1(FF / BN, M / BM);

  for (int l = 0; l < L; ++l) {
    const size_t dd = (size_t)Dm * Dm, df = (size_t)Dm * FF;
    cvt_f32_bf16<<<1024, 256, 0, stream>>>(Wq + l * dd, wb, (int)dd);
    gemm_bf16_wmma<<<gProj, 256, 0, stream>>>(hb, wb, bq + l * Dm, nullptr, qb,
                                              M, Dm, Dm, 0);
    cvt_f32_bf16<<<1024, 256, 0, stream>>>(Wk + l * dd, wb, (int)dd);
    gemm_bf16_wmma<<<gProj, 256, 0, stream>>>(hb, wb, bk + l * Dm, nullptr, kb,
                                              M, Dm, Dm, 0);
    cvt_f32_bf16<<<1024, 256, 0, stream>>>(Wv + l * dd, wb, (int)dd);
    gemm_bf16_wmma<<<gProj, 256, 0, stream>>>(hb, wb, bv + l * Dm, nullptr, vb,
                                              M, Dm, Dm, 0);
    flash_attn<<<dim3(S / 64, B * H), 128, 0, stream>>>(qb, kb, vb, mask, aob,
                                                        B, S, Dm, H);
    cvt_f32_bf16<<<1024, 256, 0, stream>>>(Wo + l * dd, wb, (int)dd);
    gemm_bf16_wmma<<<gProj, 256, 0, stream>>>(aob, wb, bo + l * Dm, tf, nullptr,
                                              M, Dm, Dm, 0);
    resid_ln<<<M, 256, 0, stream>>>(h, tf, g1 + l * Dm, be1 + l * Dm, h, hb, Dm);
    cvt_f32_bf16<<<2048, 256, 0, stream>>>(W1 + l * df, wb, (int)df);
    gemm_bf16_wmma<<<gFF1, 256, 0, stream>>>(hb, wb, b1 + l * FF, nullptr, f1b,
                                             M, FF, Dm, 1);
    cvt_f32_bf16<<<2048, 256, 0, stream>>>(W2 + l * df, wb, (int)df);
    gemm_bf16_wmma<<<gProj, 256, 0, stream>>>(f1b, wb, b2 + l * Dm, tf, nullptr,
                                              M, Dm, FF, 0);
    resid_ln<<<M, 256, 0, stream>>>(h, tf, g2 + l * Dm, be2 + l * Dm, h, hb, Dm);
  }
}
